// graph12_42992622633049
// MI455X (gfx1250) — matrix operand order repaired and checked by reading, not compiled, not run on hardware
//
#include <hip/hip_runtime.h>
#include <hip/hip_bf16.h>
#include <math.h>

typedef __attribute__((ext_vector_type(16))) __bf16 v16bf;
typedef __attribute__((ext_vector_type(8)))  float  v8f;
typedef __attribute__((ext_vector_type(4)))  float  fx4;

// Explicit global-address-space accessors: guarantees global_load/global_store
// (not flat_*) regardless of address-space inference.
#define GAS __attribute__((address_space(1)))
__device__ __forceinline__ float gldf(const float* p) {
  return *(const GAS float*)p;
}
__device__ __forceinline__ fx4 gldf4(const float* p) {
  return *(const GAS fx4*)p;
}
__device__ __forceinline__ v16bf gldbf16(const __bf16* p) {
  return *(const GAS v16bf*)p;
}
__device__ __forceinline__ void gstf(float* p, float v) {
  *(GAS float*)p = v;
}

// ---------- ws (bf16 weight pack) layout, element offsets ----------
#define WS_W1  0         // [2048][64]
#define WS_W2  131072    // [64][1024]
#define WS_L1  196608    // [3072][64]  (50 cols valid, rest 0)
#define WS_L2  393216    // [64][64]    (50x50 valid)
#define WS_L3  397312    // [64][16]    (50x8 valid)
#define WS_TOT 398336

__device__ __forceinline__ v8f wmma_bf16(v16bf a, v16bf b, v8f c) {
  return __builtin_amdgcn_wmma_f32_16x16x32_bf16(false, a, false, b, (short)0, c,
                                                 false, false);
}

// B fragment for 16x16x32 bf16: lane holds col (ncol+lane%16) of W (row-major,
// ldn columns), element i at k = kbase + 8*(lane>=16) + (i<8 ? i : i+8).
__device__ __forceinline__ v16bf load_bfrag(const __bf16* __restrict__ W, int ldn,
                                            int kbase, int ncol, int lane) {
  int kb = kbase + ((lane >> 4) << 3);
  const __bf16* p = W + (size_t)kb * ldn + ncol + (lane & 15);
  v16bf b;
#pragma unroll
  for (int i = 0; i < 8; ++i) { b[i] = p[(size_t)i * ldn]; b[i + 8] = p[(size_t)(16 + i) * ldn]; }
  return b;
}

// A fragment builder from a per-element functor: element i of lane maps to
// row r=lane&15 and k = kbase + 8*(lane>=16) + (i<8 ? i : i+8).
template <typename F>
__device__ __forceinline__ v16bf make_afrag(int lane, int kbase, F f) {
  int r = lane & 15;
  int kb = kbase + ((lane >> 4) << 3);
  v16bf a;
#pragma unroll
  for (int i = 0; i < 8; ++i)  a[i] = (__bf16)f(r, kb + i);
#pragma unroll
  for (int i = 8; i < 16; ++i) a[i] = (__bf16)f(r, kb + 8 + i);
  return a;
}

// ---------- compile-time source kinds (keeps global vs LDS paths separate) ----
struct SrcX { const float*  p; };   // global x modality slice, row stride 3072
struct SrcG { const __bf16* p; };   // LDS G buffer slice,     row stride 1024

__device__ __forceinline__ float loadv(SrcX s, int r, int j) {
  return gldf(s.p + (size_t)r * 3072 + j);
}
__device__ __forceinline__ float loadv(SrcG s, int r, int j) {
  return (float)s.p[(size_t)r * 1024 + j];
}

__device__ __forceinline__ v16bf afrag(SrcX s, int lane, int kbase) {
  int r = lane & 15;
  int kb = kbase + ((lane >> 4) << 3);
  const float* p = s.p + (size_t)r * 3072 + kb;
  fx4 u0 = gldf4(p);
  fx4 u1 = gldf4(p + 4);
  fx4 u2 = gldf4(p + 16);
  fx4 u3 = gldf4(p + 20);
  v16bf a;
#pragma unroll
  for (int i = 0; i < 4; ++i) {
    a[i]      = (__bf16)u0[i];
    a[i + 4]  = (__bf16)u1[i];
    a[i + 8]  = (__bf16)u2[i];
    a[i + 12] = (__bf16)u3[i];
  }
  return a;
}

__device__ __forceinline__ v16bf afrag(SrcG s, int lane, int kbase) {
  int r = lane & 15;
  int kb = kbase + ((lane >> 4) << 3);
  const __bf16* p = s.p + (size_t)r * 1024 + kb;
  v16bf a;
#pragma unroll
  for (int i = 0; i < 8; ++i) { a[i] = p[i]; a[i + 8] = p[16 + i]; }
  return a;
}

// One gf() call on a 16-row tile:
//   t = lrelu0.2(P@W1a + Q@W1b + b1)   (16x64)
//   g = tanh(t@W2 + b2)                (16x1024)
//   v = lrelu0.01(rowscale * g); write to dstG, or accumulate into tri.
template <typename SP, typename SQ>
__device__ __forceinline__ void gf_pair(SP sp, SQ sq, float (*tstage)[64],
                                        const __bf16* __restrict__ W1,
                                        const __bf16* __restrict__ W2,
                                        const float* __restrict__ b1,
                                        const float* __restrict__ b2,
                                        const float* rs, __bf16* dstG, __bf16* tri,
                                        int wid, int lane) {
  // ---- layer 1: M=16, N=64 (one 16-col tile per wave), K=1024+1024 ----
  int ncol = wid * 16;
  v8f acc = {};
  for (int ks = 0; ks < 1024; ks += 32) {
    v16bf a = afrag(sp, lane, ks);
    v16bf b = load_bfrag(W1, 64, ks, ncol, lane);
    acc = wmma_bf16(a, b, acc);
    a = afrag(sq, lane, ks);
    b = load_bfrag(W1 + (size_t)1024 * 64, 64, ks, ncol, lane);
    acc = wmma_bf16(a, b, acc);
  }
  int n = ncol + (lane & 15);
  int mofs = (lane >> 4) << 3;
  {
    float bias = gldf(b1 + n);
#pragma unroll
    for (int q = 0; q < 8; ++q) {
      float v = acc[q] + bias;
      tstage[mofs + q][n] = v > 0.f ? v : 0.2f * v;
    }
  }
  __syncthreads();
  // ---- layer 2: M=16, K=64, N=1024 (wave covers n-tiles wid, wid+4, ...) ----
  for (int nt = wid; nt < 64; nt += 4) {
    int nc = nt * 16;
    v8f a2 = {};
#pragma unroll
    for (int ks = 0; ks < 64; ks += 32) {
      v16bf a = make_afrag(lane, ks, [&](int r, int k) { return tstage[r][k]; });
      v16bf b = load_bfrag(W2, 1024, ks, nc, lane);
      a2 = wmma_bf16(a, b, a2);
    }
    int n2 = nc + (lane & 15);
    float b2v = gldf(b2 + n2);
#pragma unroll
    for (int q = 0; q < 8; ++q) {
      int m = mofs + q;
      float g = tanhf(a2[q] + b2v);
      float v = rs[m] * g;
      v = v > 0.f ? v : 0.01f * v;
      if (dstG) dstG[(size_t)m * 1024 + n2] = (__bf16)v;
      else {
        size_t idx = (size_t)m * 1024 + n2;
        tri[idx] = (__bf16)((float)tri[idx] + v);
      }
    }
  }
  __syncthreads();
}

// Pair dot between two softmaxed vectors, with 8-thread/row reduction.
template <typename SP, typename SQ>
__device__ __forceinline__ void pair_dot_reduce(SP sp, SQ sq,
                                                const float* mxP, const float* mxQ,
                                                const float* ZP, const float* ZQ,
                                                float* outD, float* red,
                                                int rr_, int g_, int tid) {
  float mP = mxP[rr_], mQ = mxQ[rr_], part = 0.f;
  for (int j = g_; j < 1024; j += 8)
    part += __expf(loadv(sp, rr_, j) - mP) * __expf(loadv(sq, rr_, j) - mQ);
  red[tid] = part;
  __syncthreads();
  if (g_ == 0) {
    float sum = 0.f;
    for (int q = 0; q < 8; ++q) sum += red[rr_ * 8 + q];
    outD[rr_] = sum / (ZP[rr_] * ZQ[rr_]);
  }
  __syncthreads();
}

// ------------------- weight prep: fp32 -> padded bf16 -------------------
__global__ void mmfuse_prep(const float* __restrict__ W1, const float* __restrict__ W2,
                            const float* __restrict__ L1, const float* __restrict__ L2,
                            const float* __restrict__ L3, __bf16* __restrict__ ws) {
  int i = blockIdx.x * blockDim.x + threadIdx.x;
  if (i < 131072) { ws[WS_W1 + i] = (__bf16)gldf(W1 + i); return; }
  i -= 131072;
  if (i < 65536) { ws[WS_W2 + i] = (__bf16)gldf(W2 + i); return; }
  i -= 65536;
  if (i < 196608) {
    int r = i >> 6, c = i & 63;
    ws[WS_L1 + i] = (__bf16)((c < 50) ? gldf(L1 + r * 50 + c) : 0.f);
    return;
  }
  i -= 196608;
  if (i < 4096) {
    int r = i >> 6, c = i & 63;
    ws[WS_L2 + i] = (__bf16)((r < 50 && c < 50) ? gldf(L2 + r * 50 + c) : 0.f);
    return;
  }
  i -= 4096;
  if (i < 1024) {
    int r = i >> 4, c = i & 15;
    ws[WS_L3 + i] = (__bf16)((r < 50 && c < 8) ? gldf(L3 + r * 8 + c) : 0.f);
  }
}

// ------------------- fused forward: 16 batch rows per workgroup -------------------
__global__ __launch_bounds__(128)
void mmfuse_fwd(const float* __restrict__ x, const float* __restrict__ attW,
                const float* __restrict__ attb,
                const float* __restrict__ b1, const float* __restrict__ b2,
                const float* __restrict__ bng, const float* __restrict__ bnb,
                const float* __restrict__ l1b, const float* __restrict__ l2b,
                const float* __restrict__ l3b,
                const __bf16* __restrict__ ws, float* __restrict__ out) {
  __shared__ __bf16 G[3 * 16 * 1024];     // a_v, a_l, v_l           (96 KB)
  __shared__ __bf16 tri[16 * 1024];       // trimodal accumulator    (32 KB)
  __shared__ float  tstage[16][64];       // gf hidden tile          (4 KB)
  __shared__ __bf16 ystage[16][64];
  __shared__ __bf16 y2stage[16][64];
  __shared__ float  zstage[16][16];
  __shared__ float  red[128], red2[128];
  __shared__ float  s_mx[6][16], s_Z[6][16], s_att[3][16];
  __shared__ float  s_uni[3][16], s_bi[3][16], s_sc[3][16];
  __shared__ float  s_tri[6][16], s_D[6][16];

  const __bf16* wsW1 = ws + WS_W1;
  const __bf16* wsW2 = ws + WS_W2;
  const __bf16* wsL1 = ws + WS_L1;
  const __bf16* wsL2 = ws + WS_L2;
  const __bf16* wsL3 = ws + WS_L3;

  const int tid  = threadIdx.x;
  const int lane = tid & 31;
  const int wid  = tid >> 5;
  const int base = blockIdx.x * 16;
  const float* xt = x + (size_t)base * 3072;
  const int rr_ = tid >> 3;   // reduction row (0..15)
  const int g_  = tid & 7;    // 8 threads per row

  // =========== Phase A: attention, per-modality softmax stats, pair dots ===========
  for (int s = 0; s < 3; ++s) {
    const float* p = xt + (size_t)rr_ * 3072 + (size_t)s * 1024;
    float dsum = 0.f, mx = -3.4e38f;
    for (int j = g_; j < 1024; j += 8) {
      float v = gldf(p + j);
      dsum += v * gldf(attW + j);
      mx = fmaxf(mx, v);
    }
    red[tid] = dsum;  red2[tid] = mx;  __syncthreads();
    if (g_ == 0) {
      float t = 0.f, m = -3.4e38f;
      for (int q = 0; q < 8; ++q) {
        t += red[rr_ * 8 + q];
        m = fmaxf(m, red2[rr_ * 8 + q]);
      }
      s_att[s][rr_] = tanhf(t + gldf(attb));
      s_mx[s][rr_] = m;
    }
    __syncthreads();
    float m = s_mx[s][rr_], zs = 0.f;
    for (int j = g_; j < 1024; j += 8) zs += __expf(gldf(p + j) - m);
    red[tid] = zs;  __syncthreads();
    if (g_ == 0) {
      float t = 0.f;
      for (int q = 0; q < 8; ++q) t += red[rr_ * 8 + q];
      s_Z[s][rr_] = t;
    }
    __syncthreads();
  }
  {
    SrcX xa{xt}, xv{xt + 1024}, xl{xt + 2048};
    pair_dot_reduce(xa, xv, s_mx[0], s_mx[1], s_Z[0], s_Z[1], s_D[0], red, rr_, g_, tid);
    pair_dot_reduce(xa, xl, s_mx[0], s_mx[2], s_Z[0], s_Z[2], s_D[1], red, rr_, g_, tid);
    pair_dot_reduce(xv, xl, s_mx[1], s_mx[2], s_Z[1], s_Z[2], s_D[2], red, rr_, g_, tid);
  }
  if (tid < 16) {  // w_uni, w_bi scalars
    int r = tid;
    float a0 = s_att[0][r], a1 = s_att[1][r], a2 = s_att[2][r];
    float mm = fmaxf(a0, fmaxf(a1, a2));
    float e0 = __expf(a0 - mm), e1 = __expf(a1 - mm), e2 = __expf(a2 - mm);
    float inv = 1.f / (e0 + e1 + e2);
    float sa = e0 * inv, sv = e1 * inv, sl = e2 * inv;
    s_uni[0][r] = sa; s_uni[1][r] = sv; s_uni[2][r] = sl;
    float sav = (sa + sv) / (s_D[0][r] + 0.5f);
    float sal = (sa + sl) / (s_D[1][r] + 0.5f);
    float svl = (sl + sv) / (s_D[2][r] + 0.5f);
    s_sc[0][r] = sav; s_sc[1][r] = sal; s_sc[2][r] = svl;
    mm = fmaxf(sav, fmaxf(sal, svl));
    e0 = __expf(sav - mm); e1 = __expf(sal - mm); e2 = __expf(svl - mm);
    inv = 1.f / (e0 + e1 + e2);
    s_bi[0][r] = e0 * inv; s_bi[1][r] = e1 * inv; s_bi[2][r] = e2 * inv;
    float* wrow = out + (size_t)16384 * 8 + (size_t)(base + r) * 12;
    gstf(wrow + 0, sa);       gstf(wrow + 1, sv);       gstf(wrow + 2, sl);
    gstf(wrow + 3, e0 * inv); gstf(wrow + 4, e1 * inv); gstf(wrow + 5, e2 * inv);
  }
  for (int i = tid; i < 16 * 1024; i += 128) tri[i] = (__bf16)0.f;
  __syncthreads();

  // =========== Phase B: level-1 gf -> G buffers ===========
  {
    SrcX xa{xt}, xv{xt + 1024}, xl{xt + 2048};
    gf_pair(xa, xv, tstage, wsW1, wsW2, b1, b2, s_bi[0], G,                 nullptr, wid, lane);
    gf_pair(xa, xl, tstage, wsW1, wsW2, b1, b2, s_bi[1], G + 16 * 1024,     nullptr, wid, lane);
    gf_pair(xv, xl, tstage, wsW1, wsW2, b1, b2, s_bi[2], G + 2 * 16 * 1024, nullptr, wid, lane);
  }

  // =========== Phase C: stats on a_v/a_l/v_l, cross dots, w_tri ===========
  for (int s = 3; s < 6; ++s) {
    const __bf16* p = G + (size_t)((s - 3) * 16 + rr_) * 1024;
    float mx = -3.4e38f;
    for (int j = g_; j < 1024; j += 8) mx = fmaxf(mx, (float)p[j]);
    red[tid] = mx;  __syncthreads();
    if (g_ == 0) {
      float t = -3.4e38f;
      for (int q = 0; q < 8; ++q) t = fmaxf(t, red[rr_ * 8 + q]);
      s_mx[s][rr_] = t;
    }
    __syncthreads();
    float m = s_mx[s][rr_], zs = 0.f;
    for (int j = g_; j < 1024; j += 8) zs += __expf((float)p[j] - m);
    red[tid] = zs;  __syncthreads();
    if (g_ == 0) {
      float t = 0.f;
      for (int q = 0; q < 8; ++q) t += red[rr_ * 8 + q];
      s_Z[s][rr_] = t;
    }
    __syncthreads();
  }
  {
    // (a_v2,v_l2) (a_v2,a_l2) (a_l2,v_l2) (a_v2,l) (a_l2,v) (v_l2,a)
    SrcG g0{G}, g1{G + 16 * 1024}, g2{G + 2 * 16 * 1024};
    SrcX xa{xt}, xv{xt + 1024}, xl{xt + 2048};
    pair_dot_reduce(g0, g2, s_mx[3], s_mx[5], s_Z[3], s_Z[5], s_D[0], red, rr_, g_, tid);
    pair_dot_reduce(g0, g1, s_mx[3], s_mx[4], s_Z[3], s_Z[4], s_D[1], red, rr_, g_, tid);
    pair_dot_reduce(g1, g2, s_mx[4], s_mx[5], s_Z[4], s_Z[5], s_D[2], red, rr_, g_, tid);
    pair_dot_reduce(g0, xl, s_mx[3], s_mx[2], s_Z[3], s_Z[2], s_D[3], red, rr_, g_, tid);
    pair_dot_reduce(g1, xv, s_mx[4], s_mx[1], s_Z[4], s_Z[1], s_D[4], red, rr_, g_, tid);
    pair_dot_reduce(g2, xa, s_mx[5], s_mx[0], s_Z[5], s_Z[0], s_D[5], red, rr_, g_, tid);
  }
  if (tid < 16) {  // w_tri scalars
    int r = tid;
    float sav = s_sc[0][r], sal = s_sc[1][r], svl = s_sc[2][r];
    float sa = s_uni[0][r], sv = s_uni[1][r], sl = s_uni[2][r];
    float tl[6];
    tl[0] = (sav + svl) / (s_D[0][r] + 0.5f);
    tl[1] = (sav + sal) / (s_D[1][r] + 0.5f);
    tl[2] = (sal + svl) / (s_D[2][r] + 0.5f);
    tl[3] = (sav + sl)  / (s_D[3][r] + 0.5f);
    tl[4] = (sal + sv)  / (s_D[4][r] + 0.5f);
    tl[5] = (sa + svl)  / (s_D[5][r] + 0.5f);
    float mm = tl[0];
    for (int q = 1; q < 6; ++q) mm = fmaxf(mm, tl[q]);
    float sum = 0.f, ee[6];
    for (int q = 0; q < 6; ++q) { ee[q] = __expf(tl[q] - mm); sum += ee[q]; }
    float inv = 1.f / sum;
    float* wrow = out + (size_t)16384 * 8 + (size_t)(base + r) * 12;
    for (int q = 0; q < 6; ++q) {
      s_tri[q][r] = ee[q] * inv;
      gstf(wrow + 6 + q, ee[q] * inv);
    }
  }
  __syncthreads();

  // =========== Phase D: level-2 gf -> trimodal accumulation ===========
  {
    SrcG g0{G}, g1{G + 16 * 1024}, g2{G + 2 * 16 * 1024};
    SrcX xa{xt}, xv{xt + 1024}, xl{xt + 2048};
    gf_pair(g0, g2, tstage, wsW1, wsW2, b1, b2, s_tri[0], nullptr, tri, wid, lane);
    gf_pair(g0, g1, tstage, wsW1, wsW2, b1, b2, s_tri[1], nullptr, tri, wid, lane);
    gf_pair(g2, g1, tstage, wsW1, wsW2, b1, b2, s_tri[2], nullptr, tri, wid, lane);
    gf_pair(g0, xl, tstage, wsW1, wsW2, b1, b2, s_tri[3], nullptr, tri, wid, lane);
    gf_pair(g1, xv, tstage, wsW1, wsW2, b1, b2, s_tri[4], nullptr, tri, wid, lane);
    gf_pair(g2, xa, tstage, wsW1, wsW2, b1, b2, s_tri[5], nullptr, tri, wid, lane);
  }

  // =========== Phase E: fusion @ l1_W -> l2 -> l3 -> softmax ===========
  const float BNF = rsqrtf(1.f + 1e-5f);
  {
    int ncol = wid * 16;
    v8f acc = {};
    for (int ks = 0; ks < 3072; ks += 32) {
      v16bf a = make_afrag(lane, ks, [&](int r, int kg) -> float {
        int sec = kg >> 10, j = kg & 1023;
        float v;
        if (sec == 0) {
          const float* pr = xt + (size_t)r * 3072 + j;
          v = (s_uni[0][r] * gldf(pr) + s_uni[1][r] * gldf(pr + 1024) +
               s_uni[2][r] * gldf(pr + 2048)) * (1.f / 3.f);
        } else if (sec == 1) {
          const __bf16* q = G + (size_t)r * 1024 + j;
          v = ((float)q[0] + (float)q[16 * 1024] + (float)q[32 * 1024]) * (1.f / 3.f);
        } else {
          v = (float)tri[(size_t)r * 1024 + j] * (1.f / 6.f);
        }
        return v * (gldf(bng + kg) * BNF) + gldf(bnb + kg);
      });
      v16bf b = load_bfrag(wsL1, 64, ks, ncol, lane);
      acc = wmma_bf16(a, b, acc);
    }
    int n = ncol + (lane & 15);
    int mofs = (lane >> 4) << 3;
    float bias = (n < 50) ? gldf(l1b + n) : 0.f;
#pragma unroll
    for (int q = 0; q < 8; ++q)
      ystage[mofs + q][n] = (__bf16)((n < 50) ? tanhf(acc[q] + bias) : 0.f);
  }
  __syncthreads();
  {
    int ncol = wid * 16;
    v8f acc = {};
#pragma unroll
    for (int ks = 0; ks < 64; ks += 32) {
      v16bf a = make_afrag(lane, ks, [&](int r, int k) { return (float)ystage[r][k]; });
      v16bf b = load_bfrag(wsL2, 64, ks, ncol, lane);
      acc = wmma_bf16(a, b, acc);
    }
    int n = ncol + (lane & 15);
    int mofs = (lane >> 4) << 3;
    float bias = (n < 50) ? gldf(l2b + n) : 0.f;
#pragma unroll
    for (int q = 0; q < 8; ++q)
      y2stage[mofs + q][n] = (__bf16)((n < 50) ? tanhf(acc[q] + bias) : 0.f);
  }
  __syncthreads();
  if (wid == 0) {  // whole wave 0 active -> EXEC all ones for WMMA
    v8f acc = {};
#pragma unroll
    for (int ks = 0; ks < 64; ks += 32) {
      v16bf a = make_afrag(lane, ks, [&](int r, int k) { return (float)y2stage[r][k]; });
      v16bf b = load_bfrag(wsL3, 16, ks, 0, lane);
      acc = wmma_bf16(a, b, acc);
    }
    int n = lane & 15;
    int mofs = (lane >> 4) << 3;
#pragma unroll
    for (int q = 0; q < 8; ++q)
      zstage[mofs + q][n] = acc[q] + ((n < 8) ? gldf(l3b + n) : 0.f);
  }
  __syncthreads();
  if (tid < 16) {  // final row softmax over 8 logits
    int r = tid;
    float mm = -3.4e38f;
    for (int c = 0; c < 8; ++c) mm = fmaxf(mm, zstage[r][c]);
    float es[8], sum = 0.f;
    for (int c = 0; c < 8; ++c) { es[c] = __expf(zstage[r][c] - mm); sum += es[c]; }
    float inv = 1.f / sum;
    float* orow = out + (size_t)(base + r) * 8;
    for (int c = 0; c < 8; ++c) gstf(orow + c, es[c] * inv);
  }
}

extern "C" void kernel_launch(void* const* d_in, const int* in_sizes, int n_in,
                              void* d_out, int out_size, void* d_ws, size_t ws_size,
                              hipStream_t stream) {
  const float* x    = (const float*)d_in[0];
  const float* attW = (const float*)d_in[1];
  const float* attb = (const float*)d_in[2];
  const float* W1   = (const float*)d_in[3];
  const float* b1   = (const float*)d_in[4];
  const float* W2   = (const float*)d_in[5];
  const float* b2   = (const float*)d_in[6];
  const float* bng  = (const float*)d_in[7];
  const float* bnb  = (const float*)d_in[8];
  const float* L1   = (const float*)d_in[9];
  const float* l1b  = (const float*)d_in[10];
  const float* L2   = (const float*)d_in[11];
  const float* l2b  = (const float*)d_in[12];
  const float* L3   = (const float*)d_in[13];
  const float* l3b  = (const float*)d_in[14];
  __bf16* ws = (__bf16*)d_ws;
  float* out = (float*)d_out;

  mmfuse_prep<<<(WS_TOT + 255) / 256, 256, 0, stream>>>(W1, W2, L1, L2, L3, ws);
  mmfuse_fwd<<<16384 / 16, 128, 0, stream>>>(x, attW, attb, b1, b2, bng, bnb,
                                             l1b, l2b, l3b, ws, out);
}
